// Attention_66700842106945
// MI455X (gfx1250) — matrix operand order repaired and checked by reading, not compiled, not run on hardware
//
#include <hip/hip_runtime.h>
#include <hip/hip_bf16.h>

#define HEADS 8
#define DIMH  64
#define QDIM  512
#define SEQ   2048
#define BATCH 4
#define MTOT  (BATCH * SEQ)   // 8192

typedef __attribute__((ext_vector_type(16))) __bf16 v16bf;
typedef __attribute__((ext_vector_type(8)))  float  v8f;
typedef int i32x4_t __attribute__((vector_size(16)));

#if defined(__has_builtin)
#if __has_builtin(__builtin_amdgcn_global_load_async_to_lds_b128)
#define USE_ASYNC_LDS 1
#endif
#endif

__device__ __forceinline__ v8f vzero8() {
  v8f z = {0.f,0.f,0.f,0.f,0.f,0.f,0.f,0.f};
  return z;
}

__device__ __forceinline__ v8f wmma_bf16(v16bf a, v16bf b, v8f c) {
  // 8 args: (neg_a, A, neg_b, B, c_mod, C, reuse_a, reuse_b)
  return __builtin_amdgcn_wmma_f32_16x16x32_bf16(false, a, false, b, (short)0, c, false, false);
}

// round-to-nearest-even f32 -> bf16 (as raw u16)
__device__ __forceinline__ unsigned short f2bf(float f) {
  unsigned int u = __builtin_bit_cast(unsigned int, f);
  u += 0x7FFFu + ((u >> 16) & 1u);
  return (unsigned short)(u >> 16);
}

// build a 16-element bf16 fragment from two 16-byte LDS chunks
__device__ __forceinline__ v16bf ldfrag(const unsigned short* p0, const unsigned short* p1) {
  union { v16bf v; uint4 u[2]; } f;
  f.u[0] = *(const uint4*)p0;
  f.u[1] = *(const uint4*)p1;
  return f.v;
}

// 16-byte global -> LDS copy, async (CDNA5 GLOBAL_LOAD_ASYNC_TO_LDS_B128) if available
__device__ __forceinline__ void cp16(unsigned short* l, const unsigned short* g) {
#ifdef USE_ASYNC_LDS
  __builtin_amdgcn_global_load_async_to_lds_b128(
      (__attribute__((address_space(1))) i32x4_t*)g,
      (__attribute__((address_space(3))) i32x4_t*)l, 0, 0);
#else
  *(uint4*)l = *(const uint4*)g;
#endif
}

// wait until at most N async ops remain outstanding (in-order completion)
template<int N>
__device__ __forceinline__ void cp_wait_n() {
#ifdef USE_ASYNC_LDS
#if __has_builtin(__builtin_amdgcn_s_wait_asynccnt)
  __builtin_amdgcn_s_wait_asynccnt((short)N);
#else
  asm volatile("s_wait_asynccnt %0" :: "i"(N) : "memory");
#endif
#endif
}

// ---------------- fp32 -> bf16 elementwise (8 elems / thread) ----------------
__global__ void conv_bf16_kernel(const float* __restrict__ in,
                                 unsigned short* __restrict__ out, int n8) {
  int i = blockIdx.x * blockDim.x + threadIdx.x;
  if (i >= n8) return;
  const float4* p = (const float4*)(in + (size_t)i * 8);
  float4 a = p[0], b = p[1];
  uint4 o;
  o.x = (unsigned)f2bf(a.x) | ((unsigned)f2bf(a.y) << 16);
  o.y = (unsigned)f2bf(a.z) | ((unsigned)f2bf(a.w) << 16);
  o.z = (unsigned)f2bf(b.x) | ((unsigned)f2bf(b.y) << 16);
  o.w = (unsigned)f2bf(b.z) | ((unsigned)f2bf(b.w) << 16);
  *(uint4*)(out + (size_t)i * 8) = o;
}

// ------------- weight fp32 [K=512][N=512] -> bf16 transposed [N][K] ----------
__global__ void conv_wt_kernel(const float* __restrict__ W,
                               unsigned short* __restrict__ Wt) {
  int idx = blockIdx.x * 256 + threadIdx.x;   // 0 .. 512*512-1
  int k = idx >> 9, n = idx & 511;
  Wt[(size_t)n * 512 + k] = f2bf(W[idx]);
}

// ---------------- tiled GEMM: C[8192x512] = A[8192x512] * Wt^T ----------------
// Double-buffered async-LDS pipeline: tile i+1 streams in while tile i computes.
// MODE 0: out bf16 [B,H,S,D]   (Q and K projections)
// MODE 2: out bf16 [B,H,D,S]   (V projection, transposed for PV GEMM)
// MODE 3: out fp32 [M,N]       (final output projection)
template<int MODE>
__global__ __launch_bounds__(128, 1)
void gemm128_kernel(const unsigned short* __restrict__ A,
                    const unsigned short* __restrict__ Bt,
                    void* __restrict__ Cout) {
  __shared__ __align__(16) unsigned short As[2][128 * 40];
  __shared__ __align__(16) unsigned short Bs[2][128 * 40];
  const int tid = threadIdx.x;
  const int wave = tid >> 5, lane = tid & 31;
  const int lh = lane >> 4, l16 = lane & 15;
  const int waveM = wave >> 1, waveN = wave & 1;
  const int m0 = blockIdx.y * 128;
  const int n0 = blockIdx.x * 128;

  // issue the 8 async 16B copies for K-tile kk (4 for A-tile, 4 for B-tile)
  auto issue = [&](int kk) {
    const int buf = kk & 1, k0 = kk * 32;
    const unsigned short* ga = A + (size_t)(m0 + tid) * QDIM + k0;
    unsigned short* la = As[buf] + tid * 40;
    cp16(la,      ga);      cp16(la + 8,  ga + 8);
    cp16(la + 16, ga + 16); cp16(la + 24, ga + 24);
    const unsigned short* gb = Bt + (size_t)(n0 + tid) * QDIM + k0;
    unsigned short* lb = Bs[buf] + tid * 40;
    cp16(lb,      gb);      cp16(lb + 8,  gb + 8);
    cp16(lb + 16, gb + 16); cp16(lb + 24, gb + 24);
  };

  v8f acc[4][4];
#pragma unroll
  for (int i = 0; i < 4; ++i)
#pragma unroll
    for (int j = 0; j < 4; ++j) acc[i][j] = vzero8();

  issue(0);
  issue(1);

#pragma unroll 1
  for (int kk = 0; kk < 16; ++kk) {
    // wait until this tile's 8 ops landed; leave the prefetch (8 ops) in flight
    if (kk < 15) cp_wait_n<8>(); else cp_wait_n<0>();
    __syncthreads();

    const unsigned short* Ab = As[kk & 1];
    const unsigned short* Bb = Bs[kk & 1];
    v16bf af[4], bf[4];
#pragma unroll
    for (int mt = 0; mt < 4; ++mt) {
      const unsigned short* p = Ab + (waveM * 64 + mt * 16 + l16) * 40;
      af[mt] = ldfrag(p + lh * 8, p + 16 + lh * 8);
    }
#pragma unroll
    for (int nt = 0; nt < 4; ++nt) {
      const unsigned short* p = Bb + (waveN * 64 + nt * 16 + l16) * 40;
      bf[nt] = ldfrag(p + lh * 8, p + 16 + lh * 8);
    }
#pragma unroll
    for (int mt = 0; mt < 4; ++mt)
#pragma unroll
      for (int nt = 0; nt < 4; ++nt)
        acc[mt][nt] = wmma_bf16(af[mt], bf[nt], acc[mt][nt]);

    __syncthreads();             // all readers of buf (kk&1) are done
    if (kk < 14) issue(kk + 2);  // refill the freed buffer
  }

#pragma unroll
  for (int mt = 0; mt < 4; ++mt) {
    const int mbase = m0 + waveM * 64 + mt * 16 + lh * 8;
#pragma unroll
    for (int nt = 0; nt < 4; ++nt) {
      const int n = n0 + waveN * 64 + nt * 16 + l16;
      if (MODE == 3) {
        float* op = (float*)Cout + (size_t)mbase * QDIM + n;
#pragma unroll
        for (int r = 0; r < 8; ++r) op[(size_t)r * QDIM] = acc[mt][nt][r];
      } else {
        const int h = n >> 6, d = n & 63;
        const int b = mbase >> 11, s = mbase & (SEQ - 1);
        if (MODE == 0) {
          unsigned short* op = (unsigned short*)Cout +
              ((size_t)(b * HEADS + h) * SEQ + s) * DIMH + d;
#pragma unroll
          for (int r = 0; r < 8; ++r) op[(size_t)r * DIMH] = f2bf(acc[mt][nt][r]);
        } else {  // MODE 2: V transposed [B,H,D,S]
          unsigned short* op = (unsigned short*)Cout +
              ((size_t)(b * HEADS + h) * DIMH + d) * SEQ + s;
#pragma unroll
          for (int r = 0; r < 8; ++r) op[r] = f2bf(acc[mt][nt][r]);
        }
      }
    }
  }
}

// ------------------------- flash attention per (b,h) -------------------------
// Q,K bf16 [B,H,S,64]; Vt bf16 [B,H,64,S]; AO bf16 [B,S,H*64].
// 128 threads (4 waves), 128 query rows / block, 32 keys / iteration,
// double-buffered async K/V staging overlapped with WMMA + softmax.
__global__ __launch_bounds__(128, 1)
void attn_kernel(const unsigned short* __restrict__ Q,
                 const unsigned short* __restrict__ K,
                 const unsigned short* __restrict__ Vt,
                 unsigned short* __restrict__ AO) {
  __shared__ __align__(16) unsigned short Qs[128 * 72];
  __shared__ __align__(16) unsigned short Ks[2][32 * 72];
  __shared__ __align__(16) unsigned short Vs[2][64 * 40];
  __shared__ __align__(16) unsigned short Ps[128 * 40];

  const int tid = threadIdx.x;
  const int wave = tid >> 5, lane = tid & 31;
  const int lh = lane >> 4, l16 = lane & 15;
  const int q0 = blockIdx.x * 128;
  const int bh = blockIdx.y;  // b*HEADS + h
  const unsigned short* Qg = Q  + (size_t)bh * SEQ * DIMH;
  const unsigned short* Kg = K  + (size_t)bh * SEQ * DIMH;
  const unsigned short* Vg = Vt + (size_t)bh * DIMH * SEQ;

  // 4 async 16B copies per thread per 32-key chunk (2 for K, 2 for Vt)
  auto issueKV = [&](int it) {
    const int buf = it & 1, j0 = it * 32;
    {
      int row = tid >> 2, c = (tid & 3) * 16;
      const unsigned short* g = Kg + (size_t)(j0 + row) * DIMH + c;
      unsigned short* l = Ks[buf] + row * 72 + c;
      cp16(l, g); cp16(l + 8, g + 8);
    }
    {
      int row = tid >> 1, c = (tid & 1) * 16;
      const unsigned short* g = Vg + (size_t)row * SEQ + j0 + c;
      unsigned short* l = Vs[buf] + row * 40 + c;
      cp16(l, g); cp16(l + 8, g + 8);
    }
  };

  // stage the 128x64 Q tile once (8 async ops; retire before chunk0 by order)
  {
    const unsigned short* g = Qg + (size_t)(q0 + tid) * DIMH;
    unsigned short* l = Qs + tid * 72;
#pragma unroll
    for (int i = 0; i < 8; ++i) cp16(l + i * 8, g + i * 8);
  }
  issueKV(0);
  issueKV(1);

  v8f O[2][4];
  float m_run[2][8], l_run[2][8];
#pragma unroll
  for (int mt = 0; mt < 2; ++mt) {
#pragma unroll
    for (int nt = 0; nt < 4; ++nt) O[mt][nt] = vzero8();
#pragma unroll
    for (int r = 0; r < 8; ++r) { m_run[mt][r] = -1e30f; l_run[mt][r] = 0.f; }
  }

  const float cs = 0.125f * 1.4426950408889634f;  // scale * log2(e)

#pragma unroll 1
  for (int it = 0; it < SEQ / 32; ++it) {
    // current chunk landed (leave the 4-op prefetch of chunk it+1 in flight)
    if (it < SEQ / 32 - 1) cp_wait_n<4>(); else cp_wait_n<0>();
    __syncthreads();

    const unsigned short* Kb = Ks[it & 1];
    const unsigned short* Vb = Vs[it & 1];

    // ---- scores: S = Q @ K^T for this wave's 32 rows x 32 keys ----
    v8f sc[2][2];
#pragma unroll
    for (int mt = 0; mt < 2; ++mt)
#pragma unroll
      for (int jt = 0; jt < 2; ++jt) sc[mt][jt] = vzero8();

#pragma unroll
    for (int ks = 0; ks < 2; ++ks) {
      v16bf aq[2], bk[2];
#pragma unroll
      for (int mt = 0; mt < 2; ++mt) {
        const unsigned short* p = Qs + (wave * 32 + mt * 16 + l16) * 72 + ks * 32;
        aq[mt] = ldfrag(p + lh * 8, p + 16 + lh * 8);
      }
#pragma unroll
      for (int jt = 0; jt < 2; ++jt) {
        const unsigned short* p = Kb + (jt * 16 + l16) * 72 + ks * 32;
        bk[jt] = ldfrag(p + lh * 8, p + 16 + lh * 8);
      }
#pragma unroll
      for (int mt = 0; mt < 2; ++mt)
#pragma unroll
        for (int jt = 0; jt < 2; ++jt)
          sc[mt][jt] = wmma_bf16(aq[mt], bk[jt], sc[mt][jt]);
    }

    // ---- online softmax (base-2 domain), rescale O, stage P to LDS ----
#pragma unroll
    for (int mt = 0; mt < 2; ++mt) {
#pragma unroll
      for (int r = 0; r < 8; ++r) {
        float s0 = sc[mt][0][r] * cs;
        float s1 = sc[mt][1][r] * cs;
        float x = fmaxf(s0, s1);
        x = fmaxf(x, __shfl_xor(x, 1));
        x = fmaxf(x, __shfl_xor(x, 2));
        x = fmaxf(x, __shfl_xor(x, 4));
        x = fmaxf(x, __shfl_xor(x, 8));
        float mo = m_run[mt][r];
        float mn = fmaxf(mo, x);
        float al = exp2f(mo - mn);
        m_run[mt][r] = mn;
        float p0 = exp2f(s0 - mn), p1 = exp2f(s1 - mn);
        float rs = p0 + p1;
        rs += __shfl_xor(rs, 1);
        rs += __shfl_xor(rs, 2);
        rs += __shfl_xor(rs, 4);
        rs += __shfl_xor(rs, 8);
        l_run[mt][r] = l_run[mt][r] * al + rs;
        O[mt][0][r] *= al; O[mt][1][r] *= al;
        O[mt][2][r] *= al; O[mt][3][r] *= al;
        unsigned short* pr = Ps + (wave * 32 + mt * 16 + lh * 8 + r) * 40;
        pr[l16] = f2bf(p0);
        pr[16 + l16] = f2bf(p1);
      }
    }

    // ---- O += P @ V (one 32-wide K step) ----
    v16bf bv[4];
#pragma unroll
    for (int nt = 0; nt < 4; ++nt) {
      const unsigned short* p = Vb + (nt * 16 + l16) * 40;
      bv[nt] = ldfrag(p + lh * 8, p + 16 + lh * 8);
    }
#pragma unroll
    for (int mt = 0; mt < 2; ++mt) {
      const unsigned short* p = Ps + (wave * 32 + mt * 16 + l16) * 40;
      v16bf ap = ldfrag(p + lh * 8, p + 16 + lh * 8);
#pragma unroll
      for (int nt = 0; nt < 4; ++nt)
        O[mt][nt] = wmma_bf16(ap, bv[nt], O[mt][nt]);
    }

    __syncthreads();                            // all readers of buf (it&1) done
    if (it < SEQ / 32 - 2) issueKV(it + 2);     // refill the freed buffer
  }

  // ---- epilogue: normalize and store [B,S,H*64] bf16 ----
  const int b = bh >> 3, h = bh & 7;
#pragma unroll
  for (int mt = 0; mt < 2; ++mt) {
    const int srow = q0 + wave * 32 + mt * 16 + lh * 8;
    float inv[8];
#pragma unroll
    for (int r = 0; r < 8; ++r) inv[r] = 1.0f / l_run[mt][r];
#pragma unroll
    for (int nt = 0; nt < 4; ++nt) {
      unsigned short* op = AO + ((size_t)(b * SEQ + srow)) * QDIM + h * DIMH + nt * 16 + l16;
#pragma unroll
      for (int r = 0; r < 8; ++r)
        op[(size_t)r * QDIM] = f2bf(O[mt][nt][r] * inv[r]);
    }
  }
}

// ------------------------------- launcher -----------------------------------
extern "C" void kernel_launch(void* const* d_in, const int* in_sizes, int n_in,
                              void* d_out, int out_size, void* d_ws, size_t ws_size,
                              hipStream_t stream) {
  const float* X  = (const float*)d_in[0];
  const float* Wq = (const float*)d_in[1];
  const float* Wk = (const float*)d_in[2];
  const float* Wv = (const float*)d_in[3];
  const float* Wo = (const float*)d_in[4];

  char* ws = (char*)d_ws;
  unsigned short* Xbf = (unsigned short*)(ws + 0);          //  8 MB
  unsigned short* Wqt = (unsigned short*)(ws + 8388608);    // .5 MB
  unsigned short* Wkt = (unsigned short*)(ws + 8912896);
  unsigned short* Wvt = (unsigned short*)(ws + 9437184);
  unsigned short* Wot = (unsigned short*)(ws + 9961472);
  unsigned short* Qb  = (unsigned short*)(ws + 10485760);   //  8 MB
  unsigned short* Kb  = (unsigned short*)(ws + 18874368);   //  8 MB
  unsigned short* Vtb = (unsigned short*)(ws + 27262976);   //  8 MB
  unsigned short* AOb = (unsigned short*)(ws + 35651584);   //  8 MB

  conv_bf16_kernel<<<dim3(MTOT * QDIM / (256 * 8)), 256, 0, stream>>>(X, Xbf, MTOT * QDIM / 8);
  conv_wt_kernel<<<dim3(1024), 256, 0, stream>>>(Wq, Wqt);
  conv_wt_kernel<<<dim3(1024), 256, 0, stream>>>(Wk, Wkt);
  conv_wt_kernel<<<dim3(1024), 256, 0, stream>>>(Wv, Wvt);
  conv_wt_kernel<<<dim3(1024), 256, 0, stream>>>(Wo, Wot);

  gemm128_kernel<0><<<dim3(4, 64), 128, 0, stream>>>(Xbf, Wqt, Qb);
  gemm128_kernel<0><<<dim3(4, 64), 128, 0, stream>>>(Xbf, Wkt, Kb);
  gemm128_kernel<2><<<dim3(4, 64), 128, 0, stream>>>(Xbf, Wvt, Vtb);

  attn_kernel<<<dim3(16, 32), 128, 0, stream>>>(Qb, Kb, Vtb, AOb);

  gemm128_kernel<3><<<dim3(4, 64), 128, 0, stream>>>(AOb, Wot, d_out);
}
